// SynthesizerAttention_25692494364994
// MI455X (gfx1250) — matrix operand order repaired and checked by reading, not compiled, not run on hardware
//
#include <hip/hip_runtime.h>
#include <stdint.h>

typedef _Float16 f16;
typedef __attribute__((ext_vector_type(16))) _Float16 v16h;
typedef __attribute__((ext_vector_type(8)))  _Float16 v8h;
typedef __attribute__((ext_vector_type(8)))  float    v8f;

#define N_EMBD 512
#define N_HEAD 8
#define HEAD_DIM 64
#define SEQ_T 1024
#define BATCH 8

// ---------------------------------------------------------------------------
// WMMA fragment builders (wave32, CDNA5 ISA 7.12.2 layouts).
__device__ __forceinline__ v16h load_a_frag(const f16* tile, int ld) {
  int lane = threadIdx.x & 31;
  const f16* p = tile + (lane & 15) * ld + ((lane >> 4) << 3);
  v8h lo = *(const v8h*)p;
  v8h hi = *(const v8h*)(p + 16);
  v16h a;
#pragma unroll
  for (int e = 0; e < 8; ++e) { a[e] = lo[e]; a[e + 8] = hi[e]; }
  return a;
}

// B (KxN) from an n-major tile: element (k,n) at tile[n*ld + k].
__device__ __forceinline__ v16h load_b_frag_t(const f16* tile, int ld) {
  int lane = threadIdx.x & 31;
  const f16* p = tile + (lane & 15) * ld + ((lane >> 4) << 3);
  v8h lo = *(const v8h*)p;
  v8h hi = *(const v8h*)(p + 16);
  v16h b;
#pragma unroll
  for (int e = 0; e < 8; ++e) { b[e] = lo[e]; b[e + 8] = hi[e]; }
  return b;
}

// 16B async copy global -> LDS (per lane). Tracked by ASYNCcnt.
__device__ __forceinline__ void async_copy16(void* lds_dst, const f16* gsrc) {
  uint32_t ldsb = (uint32_t)(uintptr_t)lds_dst;
  unsigned long long ga = (unsigned long long)(uintptr_t)gsrc;
  asm volatile("global_load_async_to_lds_b128 %0, %1, off"
               :: "v"(ldsb), "v"(ga) : "memory");
}
__device__ __forceinline__ void wait_async0() {
  asm volatile("s_wait_asynccnt 0x0" ::: "memory");
}

// ---------------------------------------------------------------------------
__global__ void cvt_f32_f16(const float* __restrict__ src, f16* __restrict__ dst, int n) {
  int i = blockIdx.x * blockDim.x + threadIdx.x;
  if (i < n) dst[i] = (f16)src[i];
}

// w2 [64,1024] f32 -> w2t [1024,64] f16 (kcol-major, j contiguous)
__global__ void transpose_w2(const float* __restrict__ src, f16* __restrict__ dst) {
  int i = blockIdx.x * blockDim.x + threadIdx.x;  // over 65536
  int kcol = i >> 6, j = i & 63;
  dst[i] = (f16)src[j * 2 * N_EMBD + kcol];
}

// ---------------------------------------------------------------------------
// GEMM: out[m,n] = act( sum_k A[m,k] * W[n,k] + bias[n] )
// Tile 128(M) x 64(N), K step 32; 4 waves, each wave owns 32 rows.
// Double-buffered LDS fed by async global->LDS copies.
// mode: 0 -> f16 [M,N]; 1 -> f16 vT layout [B,H,d,T]; 2 -> f32 [M,N]
__global__ __launch_bounds__(128) void gemm_wmma_f16(
    const f16* __restrict__ A, const f16* __restrict__ W,
    const float* __restrict__ bias,
    f16* __restrict__ outh, float* __restrict__ outf,
    int M, int N, int K, int relu, int mode) {
  __shared__ __align__(16) f16 As[2][128][32];      // m-major, row 64B
  __shared__ __align__(16) f16 Bs[2][64][32 + 16];  // n-major, row 96B (64B data)

  const int m0   = blockIdx.y * 128;
  const int n0   = blockIdx.x * 64;
  const int tid  = threadIdx.x;
  const int wave = tid >> 5;
  const int lane = tid & 31;

  auto load_tile = [&](int buf, int k0) {
    // A tile: 128 rows x 64B = 512 chunks of 16B; 4 per thread
#pragma unroll
    for (int i = 0; i < 4; ++i) {
      int cid = tid + 128 * i;
      int row = cid >> 2;
      int cb  = (cid & 3) * 8;  // f16 units, 16B chunk
      async_copy16(&As[buf][row][cb], &A[(size_t)(m0 + row) * K + k0 + cb]);
    }
    // B tile: 64 rows x 64B = 256 chunks; 2 per thread
#pragma unroll
    for (int i = 0; i < 2; ++i) {
      int cid = tid + 128 * i;
      int row = cid >> 2;
      int cb  = (cid & 3) * 8;
      async_copy16(&Bs[buf][row][cb], &W[(size_t)(n0 + row) * K + k0 + cb]);
    }
  };

  v8f acc[8] = {};  // [2 row strips][4 col tiles]

  load_tile(0, 0);
  wait_async0();
  __syncthreads();

  const int nk = K / 32;
  for (int s = 0; s < nk; ++s) {
    int buf = s & 1;
    if (s + 1 < nk) load_tile(buf ^ 1, (s + 1) * 32);  // prefetch next tile

    v16h a0 = load_a_frag(&As[buf][wave * 32][0], 32);
    v16h a1 = load_a_frag(&As[buf][wave * 32 + 16][0], 32);
#pragma unroll
    for (int t = 0; t < 4; ++t) {
      v16h bf = load_b_frag_t(&Bs[buf][t * 16][0], 32 + 16);
      acc[t]     = __builtin_amdgcn_wmma_f32_16x16x32_f16(
          false, a0, false, bf, (short)0, acc[t], false, false);
      acc[4 + t] = __builtin_amdgcn_wmma_f32_16x16x32_f16(
          false, a1, false, bf, (short)0, acc[4 + t], false, false);
    }
    wait_async0();
    __syncthreads();
  }

  const int nl = lane & 15;
  const int mb = (lane >> 4) << 3;
#pragma unroll
  for (int strip = 0; strip < 2; ++strip) {
    int mbase = m0 + wave * 32 + strip * 16 + mb;
#pragma unroll
    for (int t = 0; t < 4; ++t) {
      int n = n0 + t * 16 + nl;
      float bv = bias ? bias[n] : 0.0f;
      const v8f& a = acc[strip * 4 + t];
      if (mode == 1) {
        // vT store: idx = (m>>10)<<19 | n<<10 | (m&1023); contiguous over r
        size_t base = ((size_t)(mbase >> 10) << 19) + ((size_t)n << 10) + (size_t)(mbase & 1023);
        v8h pack;
#pragma unroll
        for (int r = 0; r < 8; ++r) pack[r] = (f16)(a[r] + bv);
        *(v8h*)&outh[base] = pack;
      } else {
#pragma unroll
        for (int r = 0; r < 8; ++r) {
          int m = mbase + r;
          float v = a[r] + bv;
          if (relu) v = v > 0.0f ? v : 0.0f;
          if (mode == 2) outf[(size_t)m * N + n] = v;
          else           outh[(size_t)m * N + n] = (f16)v;
        }
      }
    }
  }
}

// ---------------------------------------------------------------------------
// Fused attention per block = (b, h, 16-query strip).
__global__ __launch_bounds__(128) void attn_fused(
    const f16* __restrict__ h1,     // [B*T, C]
    const f16* __restrict__ w2t,    // [1024, 64]  (kcol-major)
    const float* __restrict__ b2,   // [1024]
    const f16* __restrict__ vT,     // [B, H, d, T] (t contiguous)
    const float* __restrict__ relv, // [11, 64]
    f16* __restrict__ y) {          // [B*T, C]
  const int T = SEQ_T, C = N_EMBD, d = HEAD_DIM;

  __shared__ __align__(16) f16 sp[16][1024];   // scores -> exp in place (32 KB)
  __shared__ __align__(16) f16 ah[16][64];
  __shared__ float ytile[16][64];
  __shared__ float red[16][8];
  __shared__ float rowmax[16];
  __shared__ float rowsum[16];
  __shared__ float band[16][6];

  const int q0   = blockIdx.x * 16;
  const int h    = blockIdx.y;
  const int b    = blockIdx.z;
  const int tid  = threadIdx.x;
  const int wave = tid >> 5;
  const int lane = tid & 31;

  {
    int i = tid * 8;
    int q = i >> 6, j = i & 63;
    *(v8h*)&ah[q][j] = *(const v8h*)&h1[(size_t)(b * T + q0 + q) * C + h * d + j];
  }
  __syncthreads();

  // ---- scores: wave w owns key columns [w*256, w*256+256) ----
  v16h a0 = load_a_frag(&ah[0][0], 64);
  v16h a1 = load_a_frag(&ah[0][32], 64);
  const int kcol_base = wave * 256;
  for (int t = 0; t < 16; ++t) {
    int kc = kcol_base + t * 16;
    bool live = (kc <= q0 + 15);          // wave-uniform causal skip
    v8f acc = {};
    if (live) {
      v16h bf0 = load_b_frag_t(w2t + (size_t)kc * 64, 64);       // j = 0..31
      v16h bf1 = load_b_frag_t(w2t + (size_t)kc * 64 + 32, 64);  // j = 32..63
      acc = __builtin_amdgcn_wmma_f32_16x16x32_f16(false, a0, false, bf0, (short)0, acc, false, false);
      acc = __builtin_amdgcn_wmma_f32_16x16x32_f16(false, a1, false, bf1, (short)0, acc, false, false);
    }
    int nl = lane & 15, mb = (lane >> 4) << 3;
    int col = kc + nl;
    float bv = b2[col];
#pragma unroll
    for (int r = 0; r < 8; ++r) {
      int q = mb + r;
      float val = live ? (acc[r] + bv) : -60000.0f;
      if (col > q0 + q) val = -60000.0f;
      sp[q][col] = (f16)val;
    }
  }
  __syncthreads();

  // ---- softmax: row = tid/8, 128-col segment = tid%8 ----
  const int r   = tid >> 3;
  const int seg = tid & 7;
  float mx = -1e30f;
  {
    const v8h* row = (const v8h*)&sp[r][seg * 128];
#pragma unroll 4
    for (int i = 0; i < 16; ++i) {
      v8h ch = row[i];
#pragma unroll
      for (int e = 0; e < 8; ++e) mx = fmaxf(mx, (float)ch[e]);
    }
  }
  red[r][seg] = mx;
  __syncthreads();
  if (tid < 16) {
    float m = red[tid][0];
    for (int s = 1; s < 8; ++s) m = fmaxf(m, red[tid][s]);
    rowmax[tid] = m;
  }
  __syncthreads();
  float rm = rowmax[r], sum = 0.0f;
  {
    v8h* row = (v8h*)&sp[r][seg * 128];
#pragma unroll 4
    for (int i = 0; i < 16; ++i) {
      v8h ch = row[i], o;
#pragma unroll
      for (int e = 0; e < 8; ++e) {
        float ev = __expf((float)ch[e] - rm);
        sum += ev;
        o[e] = (f16)ev;
      }
      row[i] = o;   // unnormalized exp; 1/sum folded into epilogue
    }
  }
  red[r][seg] = sum;
  __syncthreads();
  if (tid < 16) {
    float s = 0.0f;
    for (int k = 0; k < 8; ++k) s += red[tid][k];
    rowsum[tid] = s;
    // rel-position band: P at 5 newest keys + residual mass -> table[0]
    int q = q0 + tid;
    float inv = 1.0f / s, tot = 0.0f;
#pragma unroll
    for (int j = 0; j < 5; ++j) {
      float p = (q - j >= 0) ? (float)sp[tid][q - j] * inv : 0.0f;
      band[tid][j] = p;
      tot += p;
    }
    band[tid][5] = fmaxf(0.0f, 1.0f - tot);
  }

  // ---- y1 = exp @ V (causal k-range only); B frags straight from vT ----
  v8f yacc = {};
  const f16* vTh = vT + ((size_t)(b * N_HEAD + h) * d + wave * 16) * T;
  const int nsteps = (q0 + 15) / 32 + 1;
  for (int s = 0; s < nsteps; ++s) {
    int kk0 = s * 32;
    v16h af = load_a_frag(&sp[0][kk0], 1024);
    v16h bf = load_b_frag_t(vTh + kk0, T);
    yacc = __builtin_amdgcn_wmma_f32_16x16x32_f16(false, af, false, bf, (short)0, yacc, false, false);
  }
  {
    int nl = lane & 15, mb = (lane >> 4) << 3;
#pragma unroll
    for (int rr = 0; rr < 8; ++rr) ytile[mb + rr][wave * 16 + nl] = yacc[rr];
  }
  __syncthreads();

  // ---- epilogue: normalize y1, add band y2, 16B store per thread ----
  {
    int i = tid * 8;
    int q = i >> 6, n0 = i & 63;
    float inv = 1.0f / rowsum[q];
    v8h pack;
#pragma unroll
    for (int e = 0; e < 8; ++e) {
      int n = n0 + e;
      float y2v = band[q][5] * relv[n];
#pragma unroll
      for (int j = 0; j < 5; ++j) y2v += band[q][j] * relv[(5 - j) * 64 + n];
      pack[e] = (f16)(ytile[q][n] * inv + y2v);
    }
    *(v8h*)&y[(size_t)(b * T + q0 + q) * C + h * d + n0] = pack;
  }
}

// ---------------------------------------------------------------------------
extern "C" void kernel_launch(void* const* d_in, const int* in_sizes, int n_in,
                              void* d_out, int out_size, void* d_ws, size_t ws_size,
                              hipStream_t stream) {
  const float* x      = (const float*)d_in[0];
  const float* w1_w   = (const float*)d_in[1];
  const float* w1_b   = (const float*)d_in[2];
  const float* w2     = (const float*)d_in[3];
  const float* b2     = (const float*)d_in[4];
  const float* v_w    = (const float*)d_in[5];
  const float* v_b    = (const float*)d_in[6];
  const float* proj_w = (const float*)d_in[7];
  const float* proj_b = (const float*)d_in[8];
  const float* relv   = (const float*)d_in[9];
  float* out = (float*)d_out;

  const size_t NX  = (size_t)BATCH * SEQ_T * N_EMBD; // 4,194,304
  const size_t NW  = (size_t)N_EMBD * N_EMBD;        // 262,144
  const size_t NW2 = (size_t)HEAD_DIM * 2 * N_EMBD;  // 65,536

  f16* x16   = (f16*)d_ws;
  f16* w1_16 = x16 + NX;
  f16* vw16  = w1_16 + NW;
  f16* pj16  = vw16 + NW;
  f16* w2t16 = pj16 + NW;    // transposed [1024, 64]
  f16* h1_16 = w2t16 + NW2;
  f16* vT16  = h1_16 + NX;   // [B, H, d, T]
  f16* y16   = vT16 + NX;

  cvt_f32_f16<<<(int)((NX + 255) / 256), 256, 0, stream>>>(x,      x16,   (int)NX);
  cvt_f32_f16<<<(int)((NW + 255) / 256), 256, 0, stream>>>(w1_w,   w1_16, (int)NW);
  cvt_f32_f16<<<(int)((NW + 255) / 256), 256, 0, stream>>>(v_w,    vw16,  (int)NW);
  cvt_f32_f16<<<(int)((NW + 255) / 256), 256, 0, stream>>>(proj_w, pj16,  (int)NW);
  transpose_w2<<<(int)((NW2 + 255) / 256), 256, 0, stream>>>(w2, w2t16);

  const int M = BATCH * SEQ_T;
  dim3 gGemm(N_EMBD / 64, M / 128);

  // h1 = relu(x @ w1^T + b1)
  gemm_wmma_f16<<<gGemm, 128, 0, stream>>>(x16, w1_16, w1_b, h1_16, nullptr,
                                           M, N_EMBD, N_EMBD, 1, 0);
  // vT = (x @ v_w^T + v_b) stored [B,H,d,T]
  gemm_wmma_f16<<<gGemm, 128, 0, stream>>>(x16, vw16, v_b, vT16, nullptr,
                                           M, N_EMBD, N_EMBD, 0, 1);

  dim3 gAttn(SEQ_T / 16, N_HEAD, BATCH);
  attn_fused<<<gAttn, 128, 0, stream>>>(h1_16, w2t16, b2, vT16, relv, y16);

  // out = y @ proj^T + proj_b (f32)
  gemm_wmma_f16<<<gGemm, 128, 0, stream>>>(y16, pj16, proj_b, nullptr, out,
                                           M, N_EMBD, N_EMBD, 0, 2);
}
